// GenCENet_11081015623739
// MI455X (gfx1250) — matrix operand order repaired and checked by reading, not compiled, not run on hardware
//
#include <hip/hip_runtime.h>
#include <math.h>

// ---------------- problem constants (match reference) ----------------
#define N_NODES   12000
#define NFEAT     1024
#define NHID      64
#define HEADS     4
#define NCLASS    16
#define NEDGE     384000
#define BATCH_SZ  3000
#define TAU_F     0.3f
#define LAM_F     0.3f
#define NEG_SLOPE 0.2f

#define F1        (HEADS * NHID)      // 256
#define ETOT      (NEDGE + N_NODES)   // 396000 (with self loops)
#define NB        (N_NODES / BATCH_SZ) // 4

// ---------------- WMMA vector types (gfx1250, wave32) ----------------
typedef __bf16       v16bf __attribute__((ext_vector_type(16)));
typedef float        v8f   __attribute__((ext_vector_type(8)));
typedef unsigned int v8u   __attribute__((ext_vector_type(8)));

// ---------------- device helpers ----------------
__device__ __forceinline__ unsigned short f2bf(float f) {
  unsigned int u = __builtin_bit_cast(unsigned int, f);
  return (unsigned short)((u + 0x7FFFu + ((u >> 16) & 1u)) >> 16); // RNE
}
// order-preserving float->uint key for atomicMax-based segment_max
__device__ __forceinline__ unsigned int fkey(float f) {
  unsigned int u = __builtin_bit_cast(unsigned int, f);
  return (u & 0x80000000u) ? ~u : (u | 0x80000000u);
}
__device__ __forceinline__ float funkey(unsigned int u) {
  u = (u & 0x80000000u) ? (u & 0x7FFFFFFFu) : ~u;
  return __builtin_bit_cast(float, u);
}
#define NEG_INF_KEY 0x007FFFFFu  // fkey(-inf)

// ---------------- fill / convert kernels ----------------
__global__ void k_fill_f32(float* p, float v, long long n) {
  long long i = (long long)blockIdx.x * blockDim.x + threadIdx.x;
  if (i < n) p[i] = v;
}
__global__ void k_fill_u32(unsigned int* p, unsigned int v, long long n) {
  long long i = (long long)blockIdx.x * blockDim.x + threadIdx.x;
  if (i < n) p[i] = v;
}
__global__ void k_fill_bias(float* out, const float* __restrict__ bias, int rows, int cols) {
  long long i = (long long)blockIdx.x * blockDim.x + threadIdx.x;
  if (i < (long long)rows * cols) out[i] = bias[i % cols];
}
__global__ void k_cvt_bf16(const float* __restrict__ in, unsigned short* __restrict__ out, long long n) {
  long long i = (long long)blockIdx.x * blockDim.x + threadIdx.x;
  if (i < n) out[i] = f2bf(in[i]);
}
// in [K, N] fp32 -> out [N, K] bf16   (so B K-pairs are contiguous dwords)
__global__ void k_cvt_transpose_bf16(const float* __restrict__ in, unsigned short* __restrict__ out,
                                     int K, int Ncols) {
  long long i = (long long)blockIdx.x * blockDim.x + threadIdx.x;
  if (i >= (long long)K * Ncols) return;
  int k = (int)(i / Ncols), n = (int)(i % Ncols);
  out[(long long)n * K + k] = f2bf(in[i]);
}

// ---------------- WMMA bf16 GEMM: C[M,N] = A[M,K] @ B[K,N] ----------------
// A row-major bf16 [M,K]; Bt = B transposed, bf16 [N,K].
// One wave -> 16x16 tile; 4 waves/block -> 16x64. M%16==0, N%64==0, K%32==0.
__global__ void k_gemm_bf16_wmma(const unsigned short* __restrict__ A,
                                 const unsigned short* __restrict__ Bt,
                                 float* __restrict__ C, int M, int N, int K) {
  const int lane = threadIdx.x & 31;
  const int wave = threadIdx.x >> 5;
  const int m0 = blockIdx.x * 16;
  const int n0 = blockIdx.y * 64 + wave * 16;
  const int half = lane >> 4;   // 0 | 1
  const int l    = lane & 15;

  // ISA 16-bit A 16x32 layout: lanes 0-15 K={0..7,16..23}, lanes 16-31 K={8..15,24..31}
  const unsigned short* arow = A  + (size_t)(m0 + l) * K;
  // B 32x16 layout (same K order as A): lanes 0-15 K={0..7,16..23}, lanes 16-31 K={8..15,24..31} (N = lane%16)
  const unsigned short* brow = Bt + (size_t)(n0 + l) * K;
  const int akoff = half * 8;
  const int bkoff = half * 8;

  v8f c = {};
  for (int k0 = 0; k0 < K; k0 += 32) {
    v8u au, bu;
#pragma unroll
    for (int r = 0; r < 4; ++r) {
      au[r]     = *(const unsigned int*)(arow + k0 + akoff + 2 * r);
      au[r + 4] = *(const unsigned int*)(arow + k0 + akoff + 16 + 2 * r);
    }
#pragma unroll
    for (int r = 0; r < 4; ++r) {
      bu[r]     = *(const unsigned int*)(brow + k0 + bkoff + 2 * r);
      bu[r + 4] = *(const unsigned int*)(brow + k0 + bkoff + 16 + 2 * r);
    }

    v16bf a = __builtin_bit_cast(v16bf, au);
    v16bf b = __builtin_bit_cast(v16bf, bu);
    // (neg_a, A, neg_b, B, c_mod, C, reuse_a, reuse_b)
    c = __builtin_amdgcn_wmma_f32_16x16x32_bf16(false, a, false, b, (short)0, c, false, false);
  }
  // C/D layout: VGPR r -> M = m0 + r + half*8; N = n0 + l
#pragma unroll
  for (int r = 0; r < 8; ++r)
    C[(size_t)(m0 + r + half * 8) * N + (n0 + l)] = c[r];
}

// ---------------- small fp32 GEMM (one thread per output) ----------------
__global__ void k_gemm_f32(const float* __restrict__ A, const float* __restrict__ B,
                           const float* __restrict__ bias, float* __restrict__ C,
                           int M, int Ncols, int K) {
  long long t = (long long)blockIdx.x * blockDim.x + threadIdx.x;
  if (t >= (long long)M * Ncols) return;
  int m = (int)(t / Ncols), n = (int)(t % Ncols);
  float s = bias ? bias[n] : 0.0f;
  const float* a = A + (long long)m * K;
  for (int k = 0; k < K; ++k) s += a[k] * B[(long long)k * Ncols + n];
  C[t] = s;
}

// ---------------- GAT attention pieces ----------------
// e_src[n,h] = sum_c h[n,h,c]*a_src[h,c]; e_dst likewise (h is PRE-bias x@W)
__global__ void k_attn_logits(const float* __restrict__ h, const float* __restrict__ a_src,
                              const float* __restrict__ a_dst, float* __restrict__ es,
                              float* __restrict__ ed, int n, int heads, int ch) {
  int idx = blockIdx.x * blockDim.x + threadIdx.x;
  if (idx >= n * heads) return;
  int node = idx / heads, hh = idx % heads;
  const float* row = h + (size_t)node * heads * ch + (size_t)hh * ch;
  const float* as_ = a_src + hh * ch;
  const float* ad_ = a_dst + hh * ch;
  float s = 0.f, d = 0.f;
  for (int c = 0; c < ch; ++c) { s += row[c] * as_[c]; d += row[c] * ad_[c]; }
  es[idx] = s; ed[idx] = d;
}

// pass 1: leaky-relu logit per (edge,head); segment max via monotone-key atomicMax
__global__ void k_edge_logits_max(const int* __restrict__ ei, int heads,
                                  const float* __restrict__ es, const float* __restrict__ ed,
                                  float* __restrict__ elog, unsigned int* __restrict__ emaxk) {
  long long t = (long long)blockIdx.x * blockDim.x + threadIdx.x;
  if (t >= (long long)ETOT * heads) return;
  int e  = (int)(t / heads), hh = (int)(t % heads);
  int src = (e < NEDGE) ? ei[e]         : (e - NEDGE);
  int dst = (e < NEDGE) ? ei[NEDGE + e] : (e - NEDGE);
  float v = es[src * heads + hh] + ed[dst * heads + hh];
  v = (v > 0.f) ? v : v * NEG_SLOPE;
  elog[t] = v;
  atomicMax(&emaxk[dst * heads + hh], fkey(v));
}

// pass 2: ee = exp(v - max); segment-sum denom
__global__ void k_edge_exp_denom(const int* __restrict__ ei, int heads,
                                 float* __restrict__ elog, const unsigned int* __restrict__ emaxk,
                                 float* __restrict__ denom) {
  long long t = (long long)blockIdx.x * blockDim.x + threadIdx.x;
  if (t >= (long long)ETOT * heads) return;
  int e  = (int)(t / heads), hh = (int)(t % heads);
  int dst = (e < NEDGE) ? ei[NEDGE + e] : (e - NEDGE);
  float ee = __expf(elog[t] - funkey(emaxk[dst * heads + hh]));
  elog[t] = ee;
  atomicAdd(&denom[dst * heads + hh], ee);
}

// pass 3: out[dst, f] += h[src, f] * alpha   (one thread per edge-channel)
__global__ void k_edge_scatter(const int* __restrict__ ei, int heads, int ch,
                               const float* __restrict__ h, const float* __restrict__ elog,
                               const float* __restrict__ denom, float* __restrict__ out) {
  const int F = heads * ch;
  long long t = (long long)blockIdx.x * blockDim.x + threadIdx.x;
  if (t >= (long long)ETOT * F) return;
  int e = (int)(t / F), f = (int)(t % F), hh = f / ch;
  int src = (e < NEDGE) ? ei[e]         : (e - NEDGE);
  int dst = (e < NEDGE) ? ei[NEDGE + e] : (e - NEDGE);
  float alpha = elog[(long long)e * heads + hh] / (denom[dst * heads + hh] + 1e-16f);
  atomicAdd(&out[(size_t)dst * F + f], h[(size_t)src * F + f] * alpha);
}

// ---------------- head: minmax norm, combine ----------------
__global__ void k_rowminmax(const float* __restrict__ in, float* __restrict__ out, int rows, int cols) {
  int r = blockIdx.x * blockDim.x + threadIdx.x;
  if (r >= rows) return;
  const float* p = in + (long long)r * cols;
  float mn = p[0], mx = p[0];
  for (int c = 1; c < cols; ++c) { mn = fminf(mn, p[c]); mx = fmaxf(mx, p[c]); }
  float inv = 4.0f / (mx - mn);
  for (int c = 0; c < cols; ++c) out[(long long)r * cols + c] = (p[c] - mn) * inv - 2.0f;
}
__global__ void k_combine(const float* __restrict__ a, const float* __restrict__ b,
                          float* __restrict__ out, long long n) {
  long long i = (long long)blockIdx.x * blockDim.x + threadIdx.x;
  if (i < n) out[i] = 0.5f * (a[i] + b[i]);
}

// ---------------- contrastive loss ----------------
// hp[i,:] = elu(relu(nx[row0+i,:]) @ pW1 + pb1) @ pW2 + pb2 ; nrm[i] = ||hp[i,:]||
__global__ void k_proj(const float* __restrict__ nx, int row0,
                       const float* __restrict__ pW1, const float* __restrict__ pb1,
                       const float* __restrict__ pW2, const float* __restrict__ pb2,
                       float* __restrict__ hp, float* __restrict__ nrm) {
  int i = blockIdx.x * blockDim.x + threadIdx.x;
  if (i >= BATCH_SZ) return;
  float z[NCLASS], tb[NCLASS];
  for (int j = 0; j < NCLASS; ++j) z[j] = fmaxf(nx[(size_t)(row0 + i) * NCLASS + j], 0.0f);
  for (int k = 0; k < NCLASS; ++k) {
    float s = pb1[k];
    for (int j = 0; j < NCLASS; ++j) s += z[j] * pW1[j * NCLASS + k];
    tb[k] = (s > 0.f) ? s : (expf(s) - 1.0f);  // ELU
  }
  float nn = 0.f;
  for (int c = 0; c < NCLASS; ++c) {
    float s = pb2[c];
    for (int k = 0; k < NCLASS; ++k) s += tb[k] * pW2[k * NCLASS + c];
    hp[(size_t)i * NCLASS + c] = s;
    nn += s * s;
  }
  nrm[i] = sqrtf(nn);
}
// one thread per row i: sim_ij = exp(dot/(n1_i n2_j)/tau); rowsum, diag direct; colsum via atomics
__global__ void k_sim(const float* __restrict__ h1p, const float* __restrict__ h2p,
                      const float* __restrict__ n1, const float* __restrict__ n2,
                      float* __restrict__ rowsum, float* __restrict__ colsum,
                      float* __restrict__ diag) {
  int i = blockIdx.x * blockDim.x + threadIdx.x;
  if (i >= BATCH_SZ) return;
  float hi[NCLASS];
  for (int c = 0; c < NCLASS; ++c) hi[c] = h1p[(size_t)i * NCLASS + c];
  const float ni = n1[i];
  float rs = 0.f, dg = 0.f;
  for (int j = 0; j < BATCH_SZ; ++j) {
    float d = 0.f;
    const float* hj = h2p + (size_t)j * NCLASS;
    for (int c = 0; c < NCLASS; ++c) d += hi[c] * hj[c];
    float s = __expf(d / (ni * n2[j]) / TAU_F);
    rs += s;
    atomicAdd(&colsum[j], s);
    if (j == i) dg = s;
  }
  rowsum[i] = rs; diag[i] = dg;
}
__global__ void k_loss(const float* __restrict__ rowsum, const float* __restrict__ colsum,
                       const float* __restrict__ diag, float* __restrict__ loss) {
  int i = blockIdx.x * blockDim.x + threadIdx.x;
  if (i >= BATCH_SZ) return;
  float lmp = -logf(diag[i] / (rowsum[i] + 1e-8f));
  float lsc = -logf(diag[i] / (colsum[i] + 1e-8f));
  atomicAdd(loss, (LAM_F * lmp + (1.0f - LAM_F) * lsc) * (1.0f / BATCH_SZ));
}
__global__ void k_copy_scalar(const float* __restrict__ src, float* __restrict__ dst) {
  if (blockIdx.x == 0 && threadIdx.x == 0) dst[0] = src[0];
}

// ---------------- host orchestration ----------------
extern "C" void kernel_launch(void* const* d_in, const int* in_sizes, int n_in,
                              void* d_out, int out_size, void* d_ws, size_t ws_size,
                              hipStream_t stream) {
  const float* x    = (const float*)d_in[0];
  const float* W1   = (const float*)d_in[1];
  const float* a1s  = (const float*)d_in[2];
  const float* a1d  = (const float*)d_in[3];
  const float* b1   = (const float*)d_in[4];
  const float* W2   = (const float*)d_in[5];
  const float* a2s  = (const float*)d_in[6];
  const float* a2d  = (const float*)d_in[7];
  const float* b2   = (const float*)d_in[8];
  const float* l1W  = (const float*)d_in[9];
  const float* l1b  = (const float*)d_in[10];
  const float* l2W  = (const float*)d_in[11];
  const float* l2b  = (const float*)d_in[12];
  const float* pW1  = (const float*)d_in[13];
  const float* pb1  = (const float*)d_in[14];
  const float* pW2  = (const float*)d_in[15];
  const float* pb2  = (const float*)d_in[16];
  const int*   ei1  = (const int*)d_in[17];
  const int*   ei2  = (const int*)d_in[18];
  float* out = (float*)d_out;

  // bump allocator over workspace (~64 MB total)
  char* base = (char*)d_ws;
  size_t off = 0;
  auto alloc = [&](size_t bytes) -> void* {
    void* p = base + off;
    off = (off + bytes + 255) & ~(size_t)255;
    return p;
  };
  unsigned short* xbf   = (unsigned short*)alloc((size_t)N_NODES * NFEAT * 2);
  unsigned short* w1t   = (unsigned short*)alloc((size_t)F1 * NFEAT * 2);
  float* h1    = (float*)alloc((size_t)N_NODES * F1 * 4);
  float* es1   = (float*)alloc((size_t)N_NODES * HEADS * 4);
  float* ed1   = (float*)alloc((size_t)N_NODES * HEADS * 4);
  float* elog  = (float*)alloc((size_t)ETOT * HEADS * 4);      // reused (heads=1) in layer 2
  unsigned int* emaxk = (unsigned int*)alloc((size_t)N_NODES * HEADS * 4);
  float* denom = (float*)alloc((size_t)N_NODES * HEADS * 4);
  float* agg1  = (float*)alloc((size_t)N_NODES * F1 * 4);
  float* h2    = (float*)alloc((size_t)N_NODES * NCLASS * 4);
  float* es2   = (float*)alloc((size_t)N_NODES * 4);
  float* ed2   = (float*)alloc((size_t)N_NODES * 4);
  unsigned int* emaxk2 = (unsigned int*)alloc((size_t)N_NODES * 4);
  float* denom2 = (float*)alloc((size_t)N_NODES * 4);
  float* agg2  = (float*)alloc((size_t)N_NODES * NCLASS * 4);
  float* hn    = (float*)alloc((size_t)N_NODES * NCLASS * 4);
  float* t48   = (float*)alloc((size_t)N_NODES * NCLASS * 3 * 4);
  float* nx1   = (float*)alloc((size_t)N_NODES * NCLASS * 4);
  float* nx2   = (float*)alloc((size_t)N_NODES * NCLASS * 4);
  float* h1p   = (float*)alloc((size_t)BATCH_SZ * NCLASS * 4);
  float* h2p   = (float*)alloc((size_t)BATCH_SZ * NCLASS * 4);
  float* n1    = (float*)alloc((size_t)BATCH_SZ * 4);
  float* n2    = (float*)alloc((size_t)BATCH_SZ * 4);
  float* rowsum = (float*)alloc((size_t)BATCH_SZ * 4);
  float* colsum = (float*)alloc((size_t)BATCH_SZ * 4);
  float* diagb  = (float*)alloc((size_t)BATCH_SZ * 4);
  float* lossa  = (float*)alloc(256);
  (void)ws_size; (void)n_in; (void)in_sizes; (void)out_size;

  auto gsz = [](long long n, int b) { return (unsigned)((n + b - 1) / b); };

  // ---- shared across branches: bf16 convert, WMMA GEMM x@W1, attention logits
  k_cvt_bf16<<<gsz((long long)N_NODES * NFEAT, 256), 256, 0, stream>>>(x, xbf, (long long)N_NODES * NFEAT);
  k_cvt_transpose_bf16<<<gsz((long long)NFEAT * F1, 256), 256, 0, stream>>>(W1, w1t, NFEAT, F1);
  k_gemm_bf16_wmma<<<dim3(N_NODES / 16, F1 / 64), 128, 0, stream>>>(xbf, w1t, h1, N_NODES, F1, NFEAT);
  k_attn_logits<<<gsz((long long)N_NODES * HEADS, 256), 256, 0, stream>>>(h1, a1s, a1d, es1, ed1,
                                                                          N_NODES, HEADS, NHID);

  // ---- per-branch pipeline
  auto run_branch = [&](const int* ei, float* nx) {
    // GAT layer 1 (h1 shared): softmax over dst + scatter
    k_fill_bias<<<gsz((long long)N_NODES * F1, 256), 256, 0, stream>>>(agg1, b1, N_NODES, F1);
    k_fill_u32<<<gsz((long long)N_NODES * HEADS, 256), 256, 0, stream>>>(emaxk, NEG_INF_KEY,
                                                                         (long long)N_NODES * HEADS);
    k_fill_f32<<<gsz((long long)N_NODES * HEADS, 256), 256, 0, stream>>>(denom, 0.0f,
                                                                         (long long)N_NODES * HEADS);
    k_edge_logits_max<<<gsz((long long)ETOT * HEADS, 256), 256, 0, stream>>>(ei, HEADS, es1, ed1, elog, emaxk);
    k_edge_exp_denom<<<gsz((long long)ETOT * HEADS, 256), 256, 0, stream>>>(ei, HEADS, elog, emaxk, denom);
    k_edge_scatter<<<gsz((long long)ETOT * F1, 256), 256, 0, stream>>>(ei, HEADS, NHID, h1, elog, denom, agg1);

    // GAT layer 2: h2 = agg1 @ W2 (pre-bias), 1 head, 16 channels
    k_gemm_f32<<<gsz((long long)N_NODES * NCLASS, 256), 256, 0, stream>>>(agg1, W2, nullptr, h2,
                                                                          N_NODES, NCLASS, F1);
    k_attn_logits<<<gsz((long long)N_NODES, 256), 256, 0, stream>>>(h2, a2s, a2d, es2, ed2,
                                                                    N_NODES, 1, NCLASS);
    k_fill_bias<<<gsz((long long)N_NODES * NCLASS, 256), 256, 0, stream>>>(agg2, b2, N_NODES, NCLASS);
    k_fill_u32<<<gsz((long long)N_NODES, 256), 256, 0, stream>>>(emaxk2, NEG_INF_KEY, (long long)N_NODES);
    k_fill_f32<<<gsz((long long)N_NODES, 256), 256, 0, stream>>>(denom2, 0.0f, (long long)N_NODES);
    k_edge_logits_max<<<gsz((long long)ETOT, 256), 256, 0, stream>>>(ei, 1, es2, ed2, elog, emaxk2);
    k_edge_exp_denom<<<gsz((long long)ETOT, 256), 256, 0, stream>>>(ei, 1, elog, emaxk2, denom2);
    k_edge_scatter<<<gsz((long long)ETOT * NCLASS, 256), 256, 0, stream>>>(ei, 1, NCLASS, h2, elog, denom2, agg2);

    // head: row minmax norm + 2-layer linear
    k_rowminmax<<<gsz((long long)N_NODES, 256), 256, 0, stream>>>(agg2, hn, N_NODES, NCLASS);
    k_gemm_f32<<<gsz((long long)N_NODES * NCLASS * 3, 256), 256, 0, stream>>>(hn, l1W, l1b, t48,
                                                                              N_NODES, NCLASS * 3, NCLASS);
    k_gemm_f32<<<gsz((long long)N_NODES * NCLASS, 256), 256, 0, stream>>>(t48, l2W, l2b, nx,
                                                                          N_NODES, NCLASS, NCLASS * 3);
  };
  run_branch(ei1, nx1);
  run_branch(ei2, nx2);

  // ---- combined output
  k_combine<<<gsz((long long)N_NODES * NCLASS, 256), 256, 0, stream>>>(nx1, nx2, out,
                                                                       (long long)N_NODES * NCLASS);

  // ---- contrastive loss over NB batches
  k_fill_f32<<<1, 1, 0, stream>>>(lossa, 0.0f, 1);
  for (int b = 0; b < NB; ++b) {
    int row0 = b * BATCH_SZ;
    k_proj<<<gsz(BATCH_SZ, 256), 256, 0, stream>>>(nx1, row0, pW1, pb1, pW2, pb2, h1p, n1);
    k_proj<<<gsz(BATCH_SZ, 256), 256, 0, stream>>>(nx2, row0, pW1, pb1, pW2, pb2, h2p, n2);
    k_fill_f32<<<gsz(BATCH_SZ, 256), 256, 0, stream>>>(colsum, 0.0f, BATCH_SZ);
    k_sim<<<gsz(BATCH_SZ, 256), 256, 0, stream>>>(h1p, h2p, n1, n2, rowsum, colsum, diagb);
    k_loss<<<gsz(BATCH_SZ, 256), 256, 0, stream>>>(rowsum, colsum, diagb, lossa);
  }
  k_copy_scalar<<<1, 32, 0, stream>>>(lossa, out + (size_t)N_NODES * NCLASS);
}
